// DPMeanClusterStep_30829275251216
// MI455X (gfx1250) — compile-verified, matrix operand order checked by reading, not run on hardware
//
#include <hip/hip_runtime.h>

typedef float v8f __attribute__((ext_vector_type(8)));
typedef __bf16 bf16x16 __attribute__((ext_vector_type(16)));
typedef unsigned int u32x4 __attribute__((ext_vector_type(4)));
typedef unsigned int u32x2 __attribute__((ext_vector_type(2)));

union Frag { u32x4 q[2]; bf16x16 v; };   // 32B: two 16B runs -> one 16x32-bf16 fragment

#define D_DIM    256
#define ROWS     32          // rows (points) per block
#define FS_STRH  264         // LDS row stride in bf16 elems: 528B/row -> 4-bank skew, 16B aligned

// ---------- bf16 split helpers (hi = RNE(x), lo = RNE(x - hi)) ----------
__device__ inline unsigned short bf16_rne(float x) {
    unsigned u = __float_as_uint(x);
    unsigned r = u + 0x7FFFu + ((u >> 16) & 1u);
    return (unsigned short)(r >> 16);
}
__device__ inline void split_bf16(float x, unsigned short& h, unsigned short& l) {
    h = bf16_rne(x);
    l = bf16_rne(x - __uint_as_float((unsigned)h << 16));
}
__device__ inline unsigned pack2(unsigned short a, unsigned short b) {
    return (unsigned)a | ((unsigned)b << 16);
}

// ---------------------------------------------------------------------------
// Kernel 1: per-centroid prep: csq[k] = |mu_k|^2 (fp32), split mu into bf16
// hi/lo planes (K x 256 each) in workspace; zero the maxDist accumulator.
// ---------------------------------------------------------------------------
__global__ void prep_kernel(const float* __restrict__ mu,
                            float* __restrict__ csq,
                            unsigned short* __restrict__ muh,
                            unsigned short* __restrict__ mul,
                            unsigned* __restrict__ out_max,
                            int K) {
    int k = blockIdx.x * blockDim.x + threadIdx.x;
    if (k == 0) *out_max = 0u;                 // distances >= 0: uint order == float order
    if (k >= K) return;
    const float4* m4 = reinterpret_cast<const float4*>(mu + (size_t)k * D_DIM);
    u32x2* oh = reinterpret_cast<u32x2*>(muh + (size_t)k * D_DIM);
    u32x2* ol = reinterpret_cast<u32x2*>(mul + (size_t)k * D_DIM);
    float s = 0.f;
#pragma unroll 4
    for (int i = 0; i < D_DIM / 4; ++i) {
        float4 v = m4[i];
        s = fmaf(v.x, v.x, s); s = fmaf(v.y, v.y, s);
        s = fmaf(v.z, v.z, s); s = fmaf(v.w, v.w, s);
        unsigned short h0, h1, h2, h3, l0, l1, l2, l3;
        split_bf16(v.x, h0, l0); split_bf16(v.y, h1, l1);
        split_bf16(v.z, h2, l2); split_bf16(v.w, h3, l3);
        u32x2 H = { pack2(h0, h1), pack2(h2, h3) };
        u32x2 L = { pack2(l0, l1), pack2(l2, l3) };
        oh[i] = H; ol[i] = L;
    }
    csq[k] = s;
}

// ---------------------------------------------------------------------------
// Kernel 2: 3xBF16-split WMMA distance + argmin (compare d^2, sqrt once).
//   block: 256 threads = 8 waves; 32 rows per block.
//   wave w: row-group (w & 1) (16 rows), K-split (w >> 1) in {0..3}.
//   A-hi fragments register-resident; A-lo streamed from LDS per tile.
// ---------------------------------------------------------------------------
__global__ void dpmeans_assign_kernel(const float* __restrict__ features,
                                      const unsigned short* __restrict__ muh,
                                      const unsigned short* __restrict__ mul,
                                      const float* __restrict__ csq,
                                      float* __restrict__ out_dist,
                                      int* __restrict__ out_idx,
                                      unsigned* __restrict__ out_max,
                                      int K) {
    __shared__ __align__(16) unsigned short FsH[ROWS * FS_STRH];
    __shared__ __align__(16) unsigned short FsL[ROWS * FS_STRH];
    __shared__ float sFsq[ROWS];
    __shared__ float sMin[4][ROWS];
    __shared__ int   sIdx[4][ROWS];

    const int tid  = threadIdx.x;
    const int lane = tid & 31;
    const int wave = tid >> 5;
    const int row0 = blockIdx.x * ROWS;

    // ---- stage + split features tile into bf16 hi/lo LDS planes ----
    {
        const float4* gf = reinterpret_cast<const float4*>(features);
        for (int i = tid; i < ROWS * (D_DIM / 4); i += blockDim.x) {
            int r  = i >> 6;
            int c4 = i & 63;
            float4 v = gf[(size_t)(row0 + r) * (D_DIM / 4) + c4];
            unsigned short h0, h1, h2, h3, l0, l1, l2, l3;
            split_bf16(v.x, h0, l0); split_bf16(v.y, h1, l1);
            split_bf16(v.z, h2, l2); split_bf16(v.w, h3, l3);
            u32x2 H = { pack2(h0, h1), pack2(h2, h3) };
            u32x2 L = { pack2(l0, l1), pack2(l2, l3) };
            *reinterpret_cast<u32x2*>(&FsH[r * FS_STRH + c4 * 4]) = H;
            *reinterpret_cast<u32x2*>(&FsL[r * FS_STRH + c4 * 4]) = L;
        }
    }

    // ---- per-row fp32 squared norms from global (rows are L2/WGP$-hot) ----
    if (tid < ROWS) {
        const float4* fr = reinterpret_cast<const float4*>(features + (size_t)(row0 + tid) * D_DIM);
        float s = 0.f;
#pragma unroll 8
        for (int i = 0; i < D_DIM / 4; ++i) {
            float4 v = fr[i];
            s = fmaf(v.x, v.x, s); s = fmaf(v.y, v.y, s);
            s = fmaf(v.z, v.z, s); s = fmaf(v.w, v.w, s);
        }
        sFsq[tid] = s;
    }
    __syncthreads();

    const int rowgroup = wave & 1;
    const int ksplit   = wave >> 1;           // 0..3
    const int rowbase  = rowgroup * 16;
    const int laneN    = lane & 15;
    const int hi       = lane >> 4;           // 16-bit A layout: lanes16-31 hold K runs +8

    // A fragment base: two 16B runs per 32-K chunk at +0 and +32B from (chunk*64B + hi*16B)
    const unsigned short* aH = &FsH[(rowbase + laneN) * FS_STRH + 8 * hi];
    const unsigned short* aL = &FsL[(rowbase + laneN) * FS_STRH + 8 * hi];

    // ---- A-hi: keep all 8 chunk fragments register-resident (64 VGPRs) ----
    Frag aHf[D_DIM / 32];
#pragma unroll
    for (int c = 0; c < D_DIM / 32; ++c) {
        aHf[c].q[0] = *reinterpret_cast<const u32x4*>(aH + c * 32);
        aHf[c].q[1] = *reinterpret_cast<const u32x4*>(aH + c * 32 + 16);
    }

    // per-slot fsq is K-invariant: hoist (C layout: VGPR j -> M = j + 8*hi)
    float fs[8];
#pragma unroll
    for (int j = 0; j < 8; ++j) fs[j] = sFsq[rowbase + j + 8 * hi];

    float runMin[8];                          // running min of d^2
    int   runIdx[8];
#pragma unroll
    for (int j = 0; j < 8; ++j) { runMin[j] = __builtin_inff(); runIdx[j] = 0; }

    const int nKT = K >> 4;                   // 64 column tiles
    for (int kt = ksplit; kt < nKT; kt += 4) {
        // Block LICM: A-lo LDS loads must not be hoisted out (caused scratch
        // spills when the compiler tried to keep 128 VGPRs of A resident).
        asm volatile("" ::: "memory");

        const int kb  = kt << 4;
        const int col = kb + laneN;
        const unsigned short* bH = muh + (size_t)col * D_DIM + 8 * hi;
        const unsigned short* bL = mul + (size_t)col * D_DIM + 8 * hi;

        if (kt + 4 < nKT) {                   // pull next tile's centroid rows toward the WGP
            __builtin_prefetch(muh + (size_t)(kb + 64 + laneN) * D_DIM, 0, 0);
            __builtin_prefetch(mul + (size_t)(kb + 64 + laneN) * D_DIM, 0, 0);
        }

        v8f acc = {};
#pragma unroll
        for (int c = 0; c < D_DIM / 32; ++c) {        // 8 chunks of K=32
            Frag al, bh, bl;
            al.q[0] = *reinterpret_cast<const u32x4*>(aL + c * 32);        // ds_load_b128
            al.q[1] = *reinterpret_cast<const u32x4*>(aL + c * 32 + 16);   // ds_load_b128
            bh.q[0] = *reinterpret_cast<const u32x4*>(bH + c * 32);        // global_load_b128
            bh.q[1] = *reinterpret_cast<const u32x4*>(bH + c * 32 + 16);
            bl.q[0] = *reinterpret_cast<const u32x4*>(bL + c * 32);
            bl.q[1] = *reinterpret_cast<const u32x4*>(bL + c * 32 + 16);
            // dot ~= hi*hi + hi*lo + lo*hi  (fp32 accumulate; lo*lo ~2^-16 dropped)
            acc = __builtin_amdgcn_wmma_f32_16x16x32_bf16(false, aHf[c].v, false, bh.v,
                                                          (short)0, acc, false, false);
            acc = __builtin_amdgcn_wmma_f32_16x16x32_bf16(false, aHf[c].v, false, bl.v,
                                                          (short)0, acc, false, false);
            acc = __builtin_amdgcn_wmma_f32_16x16x32_bf16(false, al.v,     false, bh.v,
                                                          (short)0, acc, false, false);
        }

        // epilogue on d^2 only (sqrt deferred: monotonic)
        const float cs = csq[col];
#pragma unroll
        for (int j = 0; j < 8; ++j) {
            float d2 = fmaxf(fmaf(-2.f, acc[j], fs[j] + cs), 0.f);
            if (d2 < runMin[j]) { runMin[j] = d2; runIdx[j] = col; }
            // columns per slot strictly increase -> '<' keeps first index (argmin tie rule)
        }
    }

    // ---- butterfly min-reduce over the 16 N-lanes in each half ----
#pragma unroll
    for (int j = 0; j < 8; ++j) {
#pragma unroll
        for (int mask = 1; mask <= 8; mask <<= 1) {
            float ov = __shfl_xor(runMin[j], mask, 32);
            int   oi = __shfl_xor(runIdx[j], mask, 32);
            if (ov < runMin[j] || (ov == runMin[j] && oi < runIdx[j])) {
                runMin[j] = ov; runIdx[j] = oi;
            }
        }
    }
    if (laneN == 0) {
#pragma unroll
        for (int j = 0; j < 8; ++j) {
            const int r = rowbase + j + 8 * hi;
            sMin[ksplit][r] = runMin[j];
            sIdx[ksplit][r] = runIdx[j];
        }
    }
    __syncthreads();

    // ---- combine K-splits, single sqrt per row, block max -> global atomic ----
    if (tid < ROWS) {
        float best = sMin[0][tid];
        int   bi   = sIdx[0][tid];
#pragma unroll
        for (int s = 1; s < 4; ++s) {
            float v = sMin[s][tid];
            int   i = sIdx[s][tid];
            if (v < best || (v == best && i < bi)) { best = v; bi = i; }
        }
        float dist = __builtin_sqrtf(best);     // the only sqrt per point
        out_dist[row0 + tid] = dist;
        out_idx[row0 + tid]  = bi;

        float bm = dist;
#pragma unroll
        for (int mask = 16; mask >= 1; mask >>= 1)
            bm = fmaxf(bm, __shfl_xor(bm, mask, 32));
        if (tid == 0)
            atomicMax(out_max, __float_as_uint(bm));
    }
}

// ---------------------------------------------------------------------------
extern "C" void kernel_launch(void* const* d_in, const int* in_sizes, int n_in,
                              void* d_out, int out_size, void* d_ws, size_t ws_size,
                              hipStream_t stream) {
    const float* features = (const float*)d_in[0];   // (N,1,256) f32
    const float* mu       = (const float*)d_in[1];   // (1,K,256) f32
    const int N = in_sizes[0] / D_DIM;
    const int K = in_sizes[1] / D_DIM;

    float*    out_dist = (float*)d_out;                              // [0, N)
    int*      out_idx  = (int*)d_out + N;                            // [N, 2N) int32 bits
    unsigned* out_max  = (unsigned*)((float*)d_out + 2 * (size_t)N); // [2N] maxDist

    // workspace: csq (K f32, 4KB-aligned region) | mu_hi (K*256 bf16) | mu_lo (K*256 bf16)
    float*          csq = (float*)d_ws;
    unsigned short* muh = (unsigned short*)((char*)d_ws + 4096);
    unsigned short* mul = muh + (size_t)K * D_DIM;

    prep_kernel<<<(K + 255) / 256, 256, 0, stream>>>(mu, csq, muh, mul, out_max, K);
    dpmeans_assign_kernel<<<N / ROWS, 256, 0, stream>>>(
        features, muh, mul, csq, out_dist, out_idx, out_max, K);
}
